// QwenSelfAttention_23295902613970
// MI455X (gfx1250) — matrix operand order repaired and checked by reading, not compiled, not run on hardware
//
#include <hip/hip_runtime.h>

// ---------------------------------------------------------------------------
// Problem constants (match reference)
// ---------------------------------------------------------------------------
constexpr int B_   = 4;
constexpr int T_   = 512;
constexpr int P_   = 3584;
constexpr int S_   = 4096;      // P + T
constexpr int DM_  = 2048;      // d_model
constexpr int H_   = 16;
constexpr int KH_  = 4;
constexpr int DH_  = 128;
constexpr int NTOK = B_ * T_;   // 2048
constexpr int HD_  = H_ * DH_;  // 2048
constexpr int KD_  = KH_ * DH_; // 512

typedef __bf16 bf16_t;
typedef __attribute__((ext_vector_type(16))) __bf16 bf16x16;
typedef __attribute__((ext_vector_type(8)))  float  f32x8;

struct alignas(16) U128 { unsigned int w[4]; };
union FragU { U128 q[2]; bf16x16 v; };

#define WMMA_BF16(a, b, c) \
  __builtin_amdgcn_wmma_f32_16x16x32_bf16(false, (a), false, (b), (short)0, (c), false, false)

// ---------------------------------------------------------------------------
// helpers
// ---------------------------------------------------------------------------
__device__ __forceinline__ bf16_t f2bf(float f) {
  unsigned int u = __float_as_uint(f);
  u += 0x7FFFu + ((u >> 16) & 1u);          // round-to-nearest-even
  unsigned short s = (unsigned short)(u >> 16);
  bf16_t r;
  __builtin_memcpy(&r, &s, 2);
  return r;
}

// All buffers are < 2^31 elements: use uniform base + 32-bit lane offsets so
// the backend emits SADDR+voffset global addressing instead of 64-bit VGPR
// address pairs.

// A fragment (16x32, bf16, MxK), rows row-major with leading dim ld (elements).
// ISA 7.12.2: lanes 0-15 row m=lane, K = k0+0..7 (v0-3) and k0+16..23 (v4-7);
// lanes 16-31 K ranges shifted by +8.
__device__ __forceinline__ bf16x16 load_a_frag(const bf16_t* Abase, int ld, int m0, int k0) {
  int lane  = threadIdx.x & 31;
  int m     = m0 + (lane & 15);
  int kbase = k0 + ((lane >> 4) << 3);
  const bf16_t* p = Abase + (unsigned)(m * ld + kbase);
  FragU u;
  u.q[0] = *(const U128*)(p);
  u.q[1] = *(const U128*)(p + 16);
  return u.v;
}

// B fragment (32x16, bf16, KxN) where B[k][n] = W[n][k], W row-major [N, K].
// Lane holds column n = n0 + lane%16; lanes 0-15 K = k0+0..7 and k0+16..23,
// lanes 16-31 K ranges shifted by +8 (same K order as the A fragment).
__device__ __forceinline__ bf16x16 load_b_frag(const bf16_t* Wbase, int ld, int n0, int k0) {
  int lane  = threadIdx.x & 31;
  int n     = n0 + (lane & 15);
  int kbase = k0 + ((lane >> 4) << 3);
  const bf16_t* p = Wbase + (unsigned)(n * ld + kbase);
  FragU u;
  u.q[0] = *(const U128*)(p);
  u.q[1] = *(const U128*)(p + 16);
  return u.v;
}

// C/D fragment (16x16 f32): lane L, VGPR i -> row m = i + 8*(L>>4), col n = L&15.
__device__ __forceinline__ void store_c_frag(float* C, int ld, int m0, int n0, const f32x8& c) {
  int lane = threadIdx.x & 31;
  int n    = n0 + (lane & 15);
  int mb   = m0 + ((lane >> 4) << 3);
  unsigned base = (unsigned)(mb * ld + n);
#pragma unroll
  for (int i = 0; i < 8; ++i) C[base + (unsigned)(i * ld)] = c[i];
}

// ---------------------------------------------------------------------------
// f32 -> bf16 elementwise convert
// ---------------------------------------------------------------------------
__global__ void k_f2bf(const float* __restrict__ in, bf16_t* __restrict__ out, long n) {
  long i = (long)blockIdx.x * blockDim.x + threadIdx.x;
  long stride = (long)gridDim.x * blockDim.x;
  for (; i < n; i += stride) out[i] = f2bf(in[i]);
}

// ---------------------------------------------------------------------------
// Generic TN GEMM: C[m,n] = sum_k A[m,k] * W[n,k]   (A:[M,K] bf16, W:[N,K] bf16,
// C:[M,N] f32). 8 waves (2x4); wave tile 32x64 (2 M-frags x 4 N-frags) =
// 8 WMMA per 6 fragment loads. Block tile 64x256.
// M % 64 == 0, N % 256 == 0, K % 32 == 0.
// ---------------------------------------------------------------------------
__global__ __launch_bounds__(256) void k_gemm(const bf16_t* __restrict__ A,
                                              const bf16_t* __restrict__ W,
                                              float* __restrict__ C,
                                              int M, int Nout, int K) {
  int wave   = threadIdx.x >> 5;
  int m_base = blockIdx.x * 64  + (wave >> 2) * 32;
  int n_base = blockIdx.y * 256 + (wave & 3) * 64;
  (void)M;
  f32x8 z = {0.f, 0.f, 0.f, 0.f, 0.f, 0.f, 0.f, 0.f};
  f32x8 acc[2][4];
#pragma unroll
  for (int i = 0; i < 2; ++i)
#pragma unroll
    for (int j = 0; j < 4; ++j) acc[i][j] = z;

  for (int k0 = 0; k0 < K; k0 += 32) {
    bf16x16 fa[2], fb[4];
#pragma unroll
    for (int i = 0; i < 2; ++i) fa[i] = load_a_frag(A, K, m_base + 16 * i, k0);
#pragma unroll
    for (int j = 0; j < 4; ++j) fb[j] = load_b_frag(W, K, n_base + 16 * j, k0);
#pragma unroll
    for (int i = 0; i < 2; ++i)
#pragma unroll
      for (int j = 0; j < 4; ++j) acc[i][j] = WMMA_BF16(fa[i], fb[j], acc[i][j]);
  }
#pragma unroll
  for (int i = 0; i < 2; ++i)
#pragma unroll
    for (int j = 0; j < 4; ++j)
      store_c_frag(C, Nout, m_base + 16 * i, n_base + 16 * j, acc[i][j]);
}

// ---------------------------------------------------------------------------
// Per-head RMSNorm + RoPE, one wave per (token, head). Writes bf16 into
// out + ((b*heads + h)*outLD + outOff + t)*DH, scaled by `scale`.
//   q: heads=H,  outLD=T, outOff=0, scale = log2(e)/sqrt(DH)
//   k: heads=KH, outLD=S, outOff=P, scale = 1
// ---------------------------------------------------------------------------
__global__ __launch_bounds__(256) void k_norm_rope(const float* __restrict__ in,
                                                   const float* __restrict__ w,
                                                   const int* __restrict__ posArr,
                                                   bf16_t* __restrict__ out,
                                                   int heads, int outLD, int outOff,
                                                   float scale) {
  int gw   = (int)((blockIdx.x * blockDim.x + threadIdx.x) >> 5);
  int lane = threadIdx.x & 31;
  int n = gw / heads, h = gw % heads;
  int b = n / T_,     t = n % T_;
  const float* xp = in + (unsigned)(n * heads * DH_ + h * DH_);
  float v[4];
  float ss = 0.f;
#pragma unroll
  for (int c = 0; c < 4; ++c) { v[c] = xp[lane + 32 * c]; ss += v[c] * v[c]; }
  ss += __shfl_xor(ss, 1, 32);
  ss += __shfl_xor(ss, 2, 32);
  ss += __shfl_xor(ss, 4, 32);
  ss += __shfl_xor(ss, 8, 32);
  ss += __shfl_xor(ss, 16, 32);
  float rs = rsqrtf(ss * (1.0f / DH_) + 1e-6f);
#pragma unroll
  for (int c = 0; c < 4; ++c) v[c] = v[c] * rs * w[lane + 32 * c];
  float pos = (float)posArr[n];
  bf16_t* op = out + (unsigned)(((b * heads + h) * outLD + outOff + t) * DH_);
  const float L2BASE_OVER_HALF = 19.931568569324174f / 64.0f; // log2(1e6)/64
#pragma unroll
  for (int c = 0; c < 2; ++c) {
    int j = lane + 32 * c;
    float invf = exp2f(-(float)j * L2BASE_OVER_HALF);
    float th = pos * invf;
    float sn, cs;
    sincosf(th, &sn, &cs);
    float o1 = (cs * v[c]     - sn * v[c + 2]) * scale;  // element j
    float o2 = (cs * v[c + 2] + sn * v[c])     * scale;  // element j+64
    op[j]      = f2bf(o1);
    op[j + 64] = f2bf(o2);
  }
}

// V (new tokens): f32 [N, KH*DH] -> vT bf16 [B][KH][DH][S] at s = P + t
__global__ void k_v_transpose(const float* __restrict__ vf, bf16_t* __restrict__ vT, long count) {
  long i = (long)blockIdx.x * blockDim.x + threadIdx.x;
  long stride = (long)gridDim.x * blockDim.x;
  for (; i < count; i += stride) {
    int d  = (int)(i % DH_);
    long r = i / DH_;
    int kh = (int)(r % KH_);
    long n = r / KH_;
    int b = (int)(n / T_), t = (int)(n % T_);
    vT[(unsigned)(((b * KH_ + kh) * DH_ + d) * S_ + P_ + t)] = f2bf(vf[i]);
  }
}

// caches: k_cache/v_cache f32 [B][KH][P][DH] -> k_full rows [0,P), vT cols [0,P)
__global__ void k_cache_convert(const float* __restrict__ kc, const float* __restrict__ vc,
                                bf16_t* __restrict__ kfull, bf16_t* __restrict__ vT, long count) {
  long i = (long)blockIdx.x * blockDim.x + threadIdx.x;
  long stride = (long)gridDim.x * blockDim.x;
  for (; i < count; i += stride) {
    int d  = (int)(i % DH_);
    long r = i / DH_;
    int s  = (int)(r % P_);
    long r2 = r / P_;
    int kh = (int)(r2 % KH_);
    int b  = (int)(r2 / KH_);
    int hb = b * KH_ + kh;
    kfull[(unsigned)((hb * S_ + s) * DH_ + d)] = f2bf(kc[i]);
    vT[(unsigned)((hb * DH_ + d) * S_ + s)]    = f2bf(vc[i]);
  }
}

// ---------------------------------------------------------------------------
// Flash attention. 256 threads = 8 waves; block owns 8 consecutive 16-row
// Q tiles of ONE (b,h), and cooperatively stages each K chunk (32x128) and
// V^T chunk (128x32) into padded LDS once per 32-column step (8x less L2
// traffic than per-wave streaming). Padding: K rows 136 elems (stride 68
// dwords = 4 mod 64 banks), V rows 40 elems (stride 20 dwords) -> the 16-lane
// ds_read_b128 fragment loads are conflict-free.
// qb:   [B][H][T][DH]  bf16 (pre-scaled by log2e/sqrt(DH))
// kfull:[B][KH][S][DH] bf16
// vT:   [B][KH][DH][S] bf16
// ao:   [N_tok][H*DH]  bf16
// ---------------------------------------------------------------------------
__global__ __launch_bounds__(256) void k_flash(const bf16_t* __restrict__ qb,
                                               const bf16_t* __restrict__ kfull,
                                               const bf16_t* __restrict__ vT,
                                               bf16_t* __restrict__ ao) {
  __shared__ __align__(16) bf16_t ldsK[32][136];
  __shared__ __align__(16) bf16_t ldsV[128][40];
  __shared__ __align__(16) bf16_t pl[8][16][32];

  int wv   = threadIdx.x >> 5;
  int lane = threadIdx.x & 31;
  int gw   = blockIdx.x * 8 + wv;
  int tile    = gw & 31;       // T/16 = 32 tiles per head; 8 consecutive per block
  int headlin = gw >> 5;
  int h = headlin % H_;
  int b = headlin / H_;
  int t0 = tile << 4;
  int kh = h >> 2;             // N_REP = 4

  const bf16_t* Q  = qb    + (unsigned)(((b * H_ + h) * T_ + t0) * DH_);
  const bf16_t* Kb = kfull + (unsigned)((b * KH_ + kh) * S_ * DH_);
  const bf16_t* Vb = vT    + (unsigned)((b * KH_ + kh) * DH_ * S_);

  bf16x16 aq[4];
#pragma unroll
  for (int dc = 0; dc < 4; ++dc) aq[dc] = load_a_frag(Q, DH_, 0, dc * 32);

  f32x8 z = {0.f, 0.f, 0.f, 0.f, 0.f, 0.f, 0.f, 0.f};
  f32x8 oacc[8];
#pragma unroll
  for (int i = 0; i < 8; ++i) oacc[i] = z;
  float rmax[8], rsum[8];
#pragma unroll
  for (int i = 0; i < 8; ++i) { rmax[i] = -1e30f; rsum[i] = 0.f; }

  int rowbase = (lane >> 4) << 3;
  int colloc  = lane & 15;
  int my_s_hi = P_ + t0 + 16;                         // this wave's causal bound
  int tile_base = (blockIdx.x * 8) & 31;
  int blk_s_hi  = P_ + tile_base * 16 + 128;          // last wave's bound

  // staging assignments (uniform per thread across iterations)
  int tid  = threadIdx.x;
  int krow = tid >> 3, kseg = tid & 7;                // K: 32 rows x 8 segs x 32B
  int vrow = tid >> 1, vseg = tid & 1;                // V: 128 rows x 2 segs x 32B

  for (int s0 = 0; s0 < blk_s_hi; s0 += 32) {
    __syncthreads();   // previous iteration's LDS reads complete
    {
      const U128* gk = (const U128*)(Kb + (unsigned)((s0 + krow) * DH_ + kseg * 16));
      U128 ka = gk[0], kb2 = gk[1];
      *(U128*)(&ldsK[krow][kseg * 16])     = ka;
      *(U128*)(&ldsK[krow][kseg * 16 + 8]) = kb2;
      const U128* gv = (const U128*)(Vb + (unsigned)(vrow * S_ + s0 + vseg * 16));
      U128 va = gv[0], vb2 = gv[1];
      *(U128*)(&ldsV[vrow][vseg * 16])     = va;
      *(U128*)(&ldsV[vrow][vseg * 16 + 8]) = vb2;
    }
    __syncthreads();   // staged data visible to all waves

    if (s0 >= my_s_hi) continue;   // fully-masked for this wave; barriers done above

    // ---- scores (16 x 32 tile), K-dim = DH = 128, from LDS ----
    f32x8 sc0 = z, sc1 = z;
#pragma unroll
    for (int dc = 0; dc < 4; ++dc) {
      bf16x16 bk0 = load_b_frag(&ldsK[0][0], 136, 0,  dc * 32);
      bf16x16 bk1 = load_b_frag(&ldsK[0][0], 136, 16, dc * 32);
      sc0 = WMMA_BF16(aq[dc], bk0, sc0);
      sc1 = WMMA_BF16(aq[dc], bk1, sc1);
    }
    // ---- causal mask (boundary only): allowed if s <= P + t ----
    if (s0 + 31 > P_ + t0) {
#pragma unroll
      for (int i = 0; i < 8; ++i) {
        int lim = P_ + t0 + rowbase + i;
        if (s0 + colloc      > lim) sc0[i] = -1e30f;
        if (s0 + 16 + colloc > lim) sc1[i] = -1e30f;
      }
    }
    // ---- online softmax (log2 domain) ----
    float f[8];
#pragma unroll
    for (int i = 0; i < 8; ++i) {
      float v = fmaxf(sc0[i], sc1[i]);
      v = fmaxf(v, __shfl_xor(v, 1, 32));
      v = fmaxf(v, __shfl_xor(v, 2, 32));
      v = fmaxf(v, __shfl_xor(v, 4, 32));
      v = fmaxf(v, __shfl_xor(v, 8, 32));
      float mnew = fmaxf(rmax[i], v);
      f[i] = exp2f(rmax[i] - mnew);
      rmax[i] = mnew;
      float p0 = exp2f(sc0[i] - mnew);
      float p1 = exp2f(sc1[i] - mnew);
      sc0[i] = p0; sc1[i] = p1;
      float ps = p0 + p1;
      ps += __shfl_xor(ps, 1, 32);
      ps += __shfl_xor(ps, 2, 32);
      ps += __shfl_xor(ps, 4, 32);
      ps += __shfl_xor(ps, 8, 32);
      rsum[i] = rsum[i] * f[i] + ps;
    }
#pragma unroll
    for (int df = 0; df < 8; ++df)
#pragma unroll
      for (int i = 0; i < 8; ++i) oacc[df][i] *= f[i];
    // ---- transpose P through wave-private LDS: C-layout -> A-layout ----
#pragma unroll
    for (int i = 0; i < 8; ++i) {
      pl[wv][rowbase + i][colloc]      = f2bf(sc0[i]);
      pl[wv][rowbase + i][16 + colloc] = f2bf(sc1[i]);
    }
    asm volatile("s_wait_dscnt 0" ::: "memory");  // wave-private LDS RAW
    bf16x16 ap = load_a_frag(&pl[wv][0][0], 32, 0, 0);
    // ---- O += P @ V over 8 d-tiles, V from LDS ----
#pragma unroll
    for (int df = 0; df < 8; ++df) {
      bf16x16 bv = load_b_frag(&ldsV[0][0], 40, df * 16, 0);
      oacc[df] = WMMA_BF16(ap, bv, oacc[df]);
    }
  }
  // ---- normalize + store bf16 into [N_tok, H*DH] for the O-projection GEMM ----
  bf16_t* aop = ao + (unsigned)((b * T_ + t0) * HD_ + h * DH_);
#pragma unroll
  for (int df = 0; df < 8; ++df)
#pragma unroll
    for (int i = 0; i < 8; ++i) {
      float o = oacc[df][i] / rsum[i];
      aop[(unsigned)((rowbase + i) * HD_ + df * 16 + colloc)] = f2bf(o);
    }
}

// ---------------------------------------------------------------------------
// host launch
// ---------------------------------------------------------------------------
extern "C" void kernel_launch(void* const* d_in, const int* in_sizes, int n_in,
                              void* d_out, int out_size, void* d_ws, size_t ws_size,
                              hipStream_t stream) {
  (void)in_sizes; (void)n_in; (void)out_size; (void)ws_size;
  const float* x        = (const float*)d_in[0];
  const float* k_cache  = (const float*)d_in[1];
  const float* v_cache  = (const float*)d_in[2];
  const float* w_q      = (const float*)d_in[3];
  const float* w_k      = (const float*)d_in[4];
  const float* w_v      = (const float*)d_in[5];
  const float* w_o      = (const float*)d_in[6];
  const float* q_norm_w = (const float*)d_in[7];
  const float* k_norm_w = (const float*)d_in[8];
  const int*   positions= (const int*)d_in[9];
  float* out = (float*)d_out;

  unsigned char* ws = (unsigned char*)d_ws;
  size_t off = 0;
  auto take = [&](size_t bytes) -> void* {
    void* p = ws + off;
    off += (bytes + 255) & ~(size_t)255;
    return p;
  };
  bf16_t* xb    = (bf16_t*)take((size_t)NTOK * DM_ * 2);
  bf16_t* wqb   = (bf16_t*)take((size_t)HD_ * DM_ * 2);
  bf16_t* wkb   = (bf16_t*)take((size_t)KD_ * DM_ * 2);
  bf16_t* wvb   = (bf16_t*)take((size_t)KD_ * DM_ * 2);
  bf16_t* wob   = (bf16_t*)take((size_t)DM_ * HD_ * 2);
  float*  qf    = (float*) take((size_t)NTOK * HD_ * 4);
  float*  kf    = (float*) take((size_t)NTOK * KD_ * 4);
  float*  vf    = (float*) take((size_t)NTOK * KD_ * 4);
  bf16_t* qb    = (bf16_t*)take((size_t)B_ * H_ * T_ * DH_ * 2);
  bf16_t* kfull = (bf16_t*)take((size_t)B_ * KH_ * S_ * DH_ * 2);
  bf16_t* vT    = (bf16_t*)take((size_t)B_ * KH_ * DH_ * S_ * 2);
  bf16_t* ao    = (bf16_t*)take((size_t)NTOK * HD_ * 2);

  auto cvt = [&](const float* src, bf16_t* dst, long n) {
    int blocks = (int)((n + 255) / 256);
    k_f2bf<<<blocks, 256, 0, stream>>>(src, dst, n);
  };
  cvt(x,   xb,  (long)NTOK * DM_);
  cvt(w_q, wqb, (long)HD_ * DM_);
  cvt(w_k, wkb, (long)KD_ * DM_);
  cvt(w_v, wvb, (long)KD_ * DM_);
  cvt(w_o, wob, (long)DM_ * HD_);

  // QKV projections (block tile 64x256)
  k_gemm<<<dim3(NTOK / 64, HD_ / 256), 256, 0, stream>>>(xb, wqb, qf, NTOK, HD_, DM_);
  k_gemm<<<dim3(NTOK / 64, KD_ / 256), 256, 0, stream>>>(xb, wkb, kf, NTOK, KD_, DM_);
  k_gemm<<<dim3(NTOK / 64, KD_ / 256), 256, 0, stream>>>(xb, wvb, vf, NTOK, KD_, DM_);

  // RMSNorm + RoPE; fold softmax scale * log2(e) into q
  const float qscale = 0.08838834764831845f * 1.4426950408889634f; // 1/sqrt(128) * log2(e)
  k_norm_rope<<<(NTOK * H_) / 8, 256, 0, stream>>>(qf, q_norm_w, positions, qb, H_, T_, 0, qscale);
  k_norm_rope<<<(NTOK * KH_) / 8, 256, 0, stream>>>(kf, k_norm_w, positions, kfull, KH_, S_, P_, 1.0f);

  {
    long cnt = (long)NTOK * KD_;
    k_v_transpose<<<(int)((cnt + 255) / 256), 256, 0, stream>>>(vf, vT, cnt);
  }
  {
    long cnt = (long)B_ * KH_ * P_ * DH_;
    k_cache_convert<<<(int)((cnt + 255) / 256), 256, 0, stream>>>(k_cache, v_cache, kfull, vT, cnt);
  }

  // Flash attention: 2048 q-tiles, 8 waves/block sharing LDS-staged K/V
  k_flash<<<(B_ * H_ * (T_ / 16)) / 8, 256, 0, stream>>>(qb, kfull, vT, ao);

  // Output projection -> f32 d_out
  k_gemm<<<dim3(NTOK / 64, DM_ / 256), 256, 0, stream>>>(ao, wob, out, NTOK, DM_, HD_);
}
